// ConvModel_23261542875622
// MI455X (gfx1250) — matrix operand order repaired and checked by reading, not compiled, not run on hardware
//
#include <hip/hip_runtime.h>
#include <math.h>

// ---------------------------------------------------------------------------
// e3nn-style ConvModel on gfx1250 (MI455X).
// Per-edge-weight tensor products lifted into a shared-B GEMM:
//   m0[e,kk] = sum_{j,u} (h[e,j]*F[e,u]) * W2r[(j,u),kk] + sum_u F[e,u]*b2r[u,kk]
// -> WMMA f16 16x16x32, f32 accumulate.  A operands staged as f16 in LDS so the
// per-element A build is a single packed f16 multiply (no f32->f16 cvt chain).
// ---------------------------------------------------------------------------

typedef __attribute__((ext_vector_type(16))) _Float16 v16h;
typedef __attribute__((ext_vector_type(8)))  float    v8f;

static constexpr int kN  = 20000;
static constexpr int kE  = 150000;
static constexpr int kG  = 100;
static constexpr int kKPAD = 416;              // max padded K (17*24 -> 408 -> 416)
static constexpr float kY0       = 0.28209479177387814f;   // 1/(2*sqrt(pi))
static constexpr float kY1C      = 0.4886025119029199f;    // sqrt(3/(4*pi))
static constexpr float kInvSqrt3 = 0.5773502691896258f;
static constexpr float kPi       = 3.14159265358979323846f;

__device__ __forceinline__ float silu_f(float x) { return x / (1.f + __expf(-x)); }
__device__ __forceinline__ float sigm_f(float x) { return 1.f / (1.f + __expf(-x)); }

__device__ __forceinline__ v8f wmma_f16(v16h a, v16h b, v8f c) {
  // D = A(16x32 f16) * B(32x16 f16) + C(16x16 f32)
  return __builtin_amdgcn_wmma_f32_16x16x32_f16(false, a, false, b, (short)0, c,
                                                false, false);
}

// ---------------------------------------------------------------------------
// Kernel 1: per-edge geometry + Bessel basis (shared by all 4 layers).
// ---------------------------------------------------------------------------
__global__ void edge_prep(const int* __restrict__ ei, const float* __restrict__ pos,
                          float* __restrict__ besbuf, float4* __restrict__ e1buf) {
  int e = blockIdx.x * blockDim.x + threadIdx.x;
  if (e >= kE) return;
  int s = ei[e];
  int t = ei[kE + e];
  float dx = pos[t * 3 + 0] - pos[s * 3 + 0];
  float dy = pos[t * 3 + 1] - pos[s * 3 + 1];
  float dz = pos[t * 3 + 2] - pos[s * 3 + 2];
  float dist = sqrtf(dx * dx + dy * dy + dz * dz);
  float x = dist * 0.25f;  // dist / CUTOFF
  float env = 0.f;
  if (x < 1.f) {
    float x5 = x * x * x * x * x;
    env = 1.f / x - 28.f * x5 + 48.f * x5 * x - 21.f * x5 * x * x;
  }
#pragma unroll
  for (int k = 0; k < 10; ++k)
    besbuf[e * 12 + k] = env * __sinf((float)(k + 1) * kPi * x);
  besbuf[e * 12 + 10] = 0.f;
  besbuf[e * 12 + 11] = 0.f;
  float id = 1.f / dist;
  e1buf[e] = make_float4(kY1C * dx * id, kY1C * dy * id, kY1C * dz * id, dist);
}

// ---------------------------------------------------------------------------
// Kernel 2: node embedding lookup -> s_node[N,16].
// ---------------------------------------------------------------------------
__global__ void embed_nodes(const int* __restrict__ z, const float* __restrict__ emb,
                            float* __restrict__ s_node) {
  int idx = blockIdx.x * blockDim.x + threadIdx.x;
  if (idx >= kN * 16) return;
  s_node[idx] = emb[z[idx >> 4] * 16 + (idx & 15)];
}

// ---------------------------------------------------------------------------
// Kernel 3: build WMMA B matrices (f16, K-contiguous per output column) from
// W2[16,wn] and b2[wn].  Scalar B: [32 cols][KPAD], vector B: [16 cols][KPAD].
// K layout: K = j*U + u for j<16 (weight rows), then K = 16*U + u (bias rows).
// ---------------------------------------------------------------------------
__global__ void build_bt(const float* __restrict__ W2, const float* __restrict__ b2,
                         _Float16* __restrict__ bt_s, _Float16* __restrict__ bt_v,
                         int U, int KKs, int hasV) {
  int idx = blockIdx.x * blockDim.x + threadIdx.x;
  int wn = U * KKs + (hasV ? U * 8 : 0);
  if (idx < 32 * kKPAD) {
    int col = idx / kKPAD, K = idx % kKPAD;
    float val = 0.f;
    if (col < KKs && K < 17 * U) {
      if (K < 16 * U) val = W2[(K / U) * wn + (K % U) * KKs + col];
      else            val = b2[(K - 16 * U) * KKs + col];
    }
    bt_s[col * kKPAD + K] = (_Float16)val;
  } else {
    int j = idx - 32 * kKPAD;
    if (j < 16 * kKPAD) {
      int col = j / kKPAD, K = j % kKPAD;
      float val = 0.f;
      if (hasV && col < 8 && K < 17 * U) {
        int base = U * KKs;
        if (K < 16 * U) val = W2[(K / U) * wn + base + (K % U) * 8 + col];
        else            val = b2[base + (K - 16 * U) * 8 + col];
      }
      bt_v[col * kKPAD + K] = (_Float16)val;
    }
  }
}

// ---------------------------------------------------------------------------
// A-fragment builder: 16x32 f16 A tile for edges (rows) x lifted-K (cols).
//   lane<16 : M=lane,    halfs = K {0..15}   (+32*sstep)
//   lane>=16: M=lane-16, halfs = K {16..31}  (+32*sstep)
// value(K): K<16U -> h16[K/U]*F16[K%U]; 16U<=K<17U -> F16[K-16U] (bias); else 0
// All operands are f16 in LDS -> single packed f16 multiply per element pair.
// ---------------------------------------------------------------------------
template <int U>
__device__ __forceinline__ v16h build_a16(int sstep, int hi,
    const _Float16* __restrict__ hrow, const _Float16* __restrict__ frow) {
  v16h a;
#pragma unroll
  for (int i = 0; i < 16; ++i) {
    const int kofs = i + hi * 16;
    const int Kg = sstep * 32 + kofs;
    _Float16 val = (_Float16)0.f;
    if (Kg < 16 * U)      val = hrow[Kg / U] * frow[Kg % U];
    else if (Kg < 17 * U) val = frow[Kg - 16 * U];
    a[i] = val;
  }
  return a;
}

// ---------------------------------------------------------------------------
// Kernel 4: fused conv layer.  256 threads = 8 wave32s; each wave = 16 edges.
//   1) hidden h = silu(bessel @ W1 + b1) (VALU, stored f16)
//   2) gather node features -> f16 F tables (scalar + 3 vector components)
//   3) lifted-GEMM via WMMA f16 16x16x32 (B hoisted across vector components)
//   4) f32 atomic scatter to node accumulators
// ---------------------------------------------------------------------------
template <int U, int KK0, bool HASV>
__global__ __launch_bounds__(256) void conv_layer(
    const int* __restrict__ ei, const float* __restrict__ besbuf,
    const float4* __restrict__ e1buf, const float* __restrict__ s_in,
    const float* __restrict__ v_in, const float* __restrict__ W1g,
    const float* __restrict__ b1g, const _Float16* __restrict__ bt_s,
    const _Float16* __restrict__ bt_v, float* __restrict__ s_acc,
    float* __restrict__ v_acc, float inv) {
  constexpr int KSTEP = (17 * U + 31) / 32;  // 9 (U=16) or 13 (U=24)
  constexpr int NT = (KK0 + 15) / 16;        // scalar-output column tiles

  __shared__ _Float16 h16_s[8][16][16];
  __shared__ _Float16 fs_s[8][16][24];       // scalar-pass F table
  __shared__ _Float16 fv_s[8][16][3][24];    // vector-pass F tables (per d)
  __shared__ int      tgt_s[8][16];
  __shared__ float    W1_s[160];
  __shared__ float    b1_s[16];

  const int tid  = threadIdx.x;
  const int wave = tid >> 5;
  const int lane = tid & 31;
  const int r    = lane & 15;
  const int hi   = lane >> 4;

  for (int i = tid; i < 160; i += blockDim.x) W1_s[i] = W1g[i];
  if (tid < 16) b1_s[tid] = b1g[tid];
  __syncthreads();

  const int ntiles = (kE + 15) / 16;
  const int tile = blockIdx.x * 8 + wave;
  const bool active = tile < ntiles;

  if (active) {
    const int e = tile * 16 + r;
    // ---- hidden: each lane computes h[r][hi*8 .. hi*8+7], stores f16 ----
    float4 bq0 = *(const float4*)(besbuf + e * 12 + 0);
    float4 bq1 = *(const float4*)(besbuf + e * 12 + 4);
    float4 bq2 = *(const float4*)(besbuf + e * 12 + 8);
    float bes[10] = {bq0.x, bq0.y, bq0.z, bq0.w, bq1.x,
                     bq1.y, bq1.z, bq1.w, bq2.x, bq2.y};
    const int jb = hi * 8;
#pragma unroll
    for (int j = 0; j < 8; ++j) {
      float acc = b1_s[jb + j];
#pragma unroll
      for (int k = 0; k < 10; ++k) acc += bes[k] * W1_s[k * 16 + jb + j];
      h16_s[wave][r][jb + j] = (_Float16)silu_f(acc);
    }
    // ---- features -> f16 F tables ----
    float4 E1 = e1buf[e];
    if (hi == 0) {
      const int is = ei[e];
      tgt_s[wave][r] = ei[kE + e];
      const float4* sp = (const float4*)(s_in + is * 16);
      float sj[16];
#pragma unroll
      for (int q = 0; q < 4; ++q) {
        float4 sv = sp[q];
        sj[q * 4 + 0] = sv.x; sj[q * 4 + 1] = sv.y;
        sj[q * 4 + 2] = sv.z; sj[q * 4 + 3] = sv.w;
      }
#pragma unroll
      for (int u = 0; u < 16; ++u) fs_s[wave][r][u] = (_Float16)(sj[u] * kY0);
#pragma unroll
      for (int u = 0; u < 16; ++u) {
        fv_s[wave][r][0][u] = (_Float16)(sj[u] * E1.x);
        fv_s[wave][r][1][u] = (_Float16)(sj[u] * E1.y);
        fv_s[wave][r][2][u] = (_Float16)(sj[u] * E1.z);
      }
    } else {
      if constexpr (U > 16) {
        const int is = ei[e];
        const float4* vp = (const float4*)(v_in + is * 24);
        float vv[24];
#pragma unroll
        for (int q = 0; q < 6; ++q) {
          float4 t4 = vp[q];
          vv[q * 4 + 0] = t4.x; vv[q * 4 + 1] = t4.y;
          vv[q * 4 + 2] = t4.z; vv[q * 4 + 3] = t4.w;
        }
#pragma unroll
        for (int u = 0; u < 8; ++u) {
          fs_s[wave][r][16 + u] =
              (_Float16)((vv[u * 3] * E1.x + vv[u * 3 + 1] * E1.y +
                          vv[u * 3 + 2] * E1.z) * kInvSqrt3);
          fv_s[wave][r][0][16 + u] = (_Float16)(vv[u * 3 + 0] * kY0);
          fv_s[wave][r][1][16 + u] = (_Float16)(vv[u * 3 + 1] * kY0);
          fv_s[wave][r][2][16 + u] = (_Float16)(vv[u * 3 + 2] * kY0);
        }
      } else {
#pragma unroll
        for (int u = 0; u < 8; ++u) {
          fs_s[wave][r][16 + u] = (_Float16)0.f;
          fv_s[wave][r][0][16 + u] = (_Float16)0.f;
          fv_s[wave][r][1][16 + u] = (_Float16)0.f;
          fv_s[wave][r][2][16 + u] = (_Float16)0.f;
        }
      }
    }
  }
  __syncthreads();
  if (!active) return;

  const _Float16* hrow  = h16_s[wave][r];
  const _Float16* fsrow = fs_s[wave][r];
  const int colL = lane & 15;

  // ---------------- scalar pass: m0 [16 edges x KK0] ----------------
  {
    v8f acc0 = {};
    v8f acc1 = {};
#pragma unroll
    for (int s = 0; s < KSTEP; ++s) {
      v16h a = build_a16<U>(s, hi, hrow, fsrow);
      v16h b0 = *(const v16h*)(bt_s + colL * kKPAD + s * 32 + hi * 16);
      acc0 = wmma_f16(a, b0, acc0);
      if (NT > 1) {
        v16h b1f = *(const v16h*)(bt_s + (16 + colL) * kKPAD + s * 32 + hi * 16);
        acc1 = wmma_f16(a, b1f, acc1);
      }
    }
#pragma unroll
    for (int rI = 0; rI < 8; ++rI) {
      const int m = rI + hi * 8;          // edge row in tile (C/D layout)
      const int t = tgt_s[wave][m];
      if (colL < KK0) atomicAdd(&s_acc[t * 24 + colL], acc0[rI] * inv);
      if (NT > 1 && colL + 16 < KK0)
        atomicAdd(&s_acc[t * 24 + colL + 16], acc1[rI] * inv);
    }
  }

  // -------- vector passes: mv [16 edges x 8 x 3], B hoisted over d --------
  if constexpr (HASV) {
    v8f av0 = {};
    v8f av1 = {};
    v8f av2 = {};
#pragma unroll
    for (int s = 0; s < KSTEP; ++s) {
      v16h b = *(const v16h*)(bt_v + colL * kKPAD + s * 32 + hi * 16);
      av0 = wmma_f16(build_a16<U>(s, hi, hrow, fv_s[wave][r][0]), b, av0);
      av1 = wmma_f16(build_a16<U>(s, hi, hrow, fv_s[wave][r][1]), b, av1);
      av2 = wmma_f16(build_a16<U>(s, hi, hrow, fv_s[wave][r][2]), b, av2);
    }
    if (colL < 8) {
#pragma unroll
      for (int rI = 0; rI < 8; ++rI) {
        const int m = rI + hi * 8;
        const int t = tgt_s[wave][m];
        atomicAdd(&v_acc[t * 24 + colL * 3 + 0], av0[rI] * inv);
        atomicAdd(&v_acc[t * 24 + colL * 3 + 1], av1[rI] * inv);
        atomicAdd(&v_acc[t * 24 + colL * 3 + 2], av2[rI] * inv);
      }
    }
  }
}

// ---------------------------------------------------------------------------
// Kernel 5: gate nonlinearity (per node).
// ---------------------------------------------------------------------------
__global__ void gate_nodes(const float* __restrict__ s_acc,
                           const float* __restrict__ v_acc,
                           float* __restrict__ s_node, float* __restrict__ v_node) {
  int n = blockIdx.x * blockDim.x + threadIdx.x;
  if (n >= kN) return;
  float g[8];
#pragma unroll
  for (int k = 0; k < 8; ++k) g[k] = sigm_f(s_acc[n * 24 + k]);
#pragma unroll
  for (int m = 0; m < 16; ++m) s_node[n * 16 + m] = silu_f(s_acc[n * 24 + 8 + m]);
#pragma unroll
  for (int k = 0; k < 8; ++k)
#pragma unroll
    for (int d = 0; d < 3; ++d)
      v_node[n * 24 + k * 3 + d] = v_acc[n * 24 + k * 3 + d] * g[k];
}

// ---------------------------------------------------------------------------
// Kernel 6: global_add_pool over graphs.
// ---------------------------------------------------------------------------
__global__ void pool_nodes(const float* __restrict__ s_acc,
                           const int* __restrict__ batch, float* __restrict__ out) {
  int n = blockIdx.x * blockDim.x + threadIdx.x;
  if (n >= kN) return;
  int b = batch[n];
#pragma unroll
  for (int k = 0; k < 4; ++k) atomicAdd(&out[b * 4 + k], s_acc[n * 24 + k]);
}

// ---------------------------------------------------------------------------
extern "C" void kernel_launch(void* const* d_in, const int* in_sizes, int n_in,
                              void* d_out, int out_size, void* d_ws, size_t ws_size,
                              hipStream_t stream) {
  (void)in_sizes; (void)n_in; (void)ws_size;
  const int*   ei    = (const int*)d_in[0];     // edge_index [2,E]
  const int*   z     = (const int*)d_in[1];
  const float* pos   = (const float*)d_in[2];
  const int*   batch = (const int*)d_in[3];
  const float* emb   = (const float*)d_in[4];
  const float *W1p[4], *b1p[4], *W2p[4], *b2p[4];
  for (int li = 0; li < 4; ++li) {              // dict order: freq,W1,b1,W2,b2
    W1p[li] = (const float*)d_in[6 + 5 * li];
    b1p[li] = (const float*)d_in[7 + 5 * li];
    W2p[li] = (const float*)d_in[8 + 5 * li];
    b2p[li] = (const float*)d_in[9 + 5 * li];
  }

  char* cur = (char*)d_ws;
  auto carve = [&](size_t bytes) -> void* {
    void* p = (void*)cur;
    cur += (bytes + 255) & ~size_t(255);
    return p;
  };
  float4*   e1buf  = (float4*)carve(sizeof(float4) * kE);
  float*    bes    = (float*)carve(sizeof(float) * kE * 12);
  float*    s_node = (float*)carve(sizeof(float) * kN * 16);
  float*    v_node = (float*)carve(sizeof(float) * kN * 24);
  float*    s_acc  = (float*)carve(sizeof(float) * kN * 24);
  float*    v_acc  = (float*)carve(sizeof(float) * kN * 24);
  _Float16* bts    = (_Float16*)carve(sizeof(_Float16) * 32 * kKPAD);
  _Float16* btv    = (_Float16*)carve(sizeof(_Float16) * 16 * kKPAD);

  const float invS = 0.25f;                    // 1/sqrt(16)
  const float invU = 0.20412414523193154f;     // 1/sqrt(24)
  const int nblocks = (((kE + 15) / 16) + 7) / 8;
  const int btThreads = (48 * kKPAD + 255) / 256;

  edge_prep<<<(kE + 255) / 256, 256, 0, stream>>>(ei, pos, bes, e1buf);
  embed_nodes<<<(kN * 16 + 255) / 256, 256, 0, stream>>>(z, emb, s_node);

  // ---- layer 0: 16x0e in ----
  build_bt<<<btThreads, 256, 0, stream>>>(W2p[0], b2p[0], bts, btv, 16, 24, 1);
  hipMemsetAsync(s_acc, 0, sizeof(float) * kN * 24, stream);
  hipMemsetAsync(v_acc, 0, sizeof(float) * kN * 24, stream);
  conv_layer<16, 24, true><<<nblocks, 256, 0, stream>>>(
      ei, bes, e1buf, s_node, v_node, W1p[0], b1p[0], bts, btv, s_acc, v_acc, invS);
  gate_nodes<<<(kN + 255) / 256, 256, 0, stream>>>(s_acc, v_acc, s_node, v_node);

  // ---- layers 1,2 ----
  for (int li = 1; li <= 2; ++li) {
    build_bt<<<btThreads, 256, 0, stream>>>(W2p[li], b2p[li], bts, btv, 24, 24, 1);
    hipMemsetAsync(s_acc, 0, sizeof(float) * kN * 24, stream);
    hipMemsetAsync(v_acc, 0, sizeof(float) * kN * 24, stream);
    conv_layer<24, 24, true><<<nblocks, 256, 0, stream>>>(
        ei, bes, e1buf, s_node, v_node, W1p[li], b1p[li], bts, btv, s_acc, v_acc, invU);
    gate_nodes<<<(kN + 255) / 256, 256, 0, stream>>>(s_acc, v_acc, s_node, v_node);
  }

  // ---- layer 3: -> 4x0e, no gate ----
  build_bt<<<btThreads, 256, 0, stream>>>(W2p[3], b2p[3], bts, btv, 24, 4, 0);
  hipMemsetAsync(s_acc, 0, sizeof(float) * kN * 24, stream);
  conv_layer<24, 4, false><<<nblocks, 256, 0, stream>>>(
      ei, bes, e1buf, s_node, v_node, W1p[3], b1p[3], bts, btv, s_acc, v_acc, invU);

  hipMemsetAsync(d_out, 0, sizeof(float) * out_size, stream);
  pool_nodes<<<(kN + 255) / 256, 256, 0, stream>>>(s_acc, batch, (float*)d_out);
}
